// UltraContextModule_15805479649735
// MI455X (gfx1250) — compile-verified, matrix operand order checked by reading, not run on hardware
//
#include <hip/hip_runtime.h>

#define S_TOK 8192
#define D_MODEL 1024
#define NHEAD 16
#define HD 64
#define WIN 1024
#define G_TOK 256
#define NEGINF -1e30f

typedef __attribute__((ext_vector_type(16))) __bf16 v16bf;
typedef __attribute__((ext_vector_type(8)))  float  v8f;

static __device__ __forceinline__ unsigned short f2bf(float f) {
    unsigned int u = __float_as_uint(f);
    u += 0x7FFFu + ((u >> 16) & 1u);   // round-to-nearest-even
    return (unsigned short)(u >> 16);
}

// Load a 16x32 bf16 A/B fragment (wave32 layout) from a row-major matrix.
// lane<16: row=row0+lane, k = k0 + {0..7,16..23}; lane>=16: row=row0+lane-16, k = k0+{8..15,24..31}
static __device__ __forceinline__ v16bf load_frag_g(const unsigned short* __restrict__ base,
                                                    int ld, int row0, int k0) {
    int lane = threadIdx.x & 31;
    int r = row0 + (lane & 15);
    int kk = k0 + ((lane & 16) >> 1);      // +8 for upper half-wave
    const unsigned short* p = base + (size_t)r * ld + kk;
    union { uint4 q[2]; v16bf v; } u;
    u.q[0] = *(const uint4*)(p);
    u.q[1] = *(const uint4*)(p + 16);
    return u.v;
}

// ---------------- Kernel 1: RMSNorm + scorer ----------------
__global__ __launch_bounds__(256)
void rmsnorm_score_kernel(const float* __restrict__ x, const float* __restrict__ norm_w,
                          const float* __restrict__ scorer_w, const float* __restrict__ scorer_b,
                          unsigned short* __restrict__ h_bf, float* __restrict__ scores) {
    __shared__ float red[256];
    const int row = blockIdx.x, tid = threadIdx.x;
    const float* xr = x + (size_t)row * D_MODEL;
    float xv[4];
    float ssq = 0.f;
#pragma unroll
    for (int j = 0; j < 4; ++j) { xv[j] = xr[tid + j * 256]; ssq += xv[j] * xv[j]; }
    red[tid] = ssq; __syncthreads();
    for (int st = 128; st > 0; st >>= 1) { if (tid < st) red[tid] += red[tid + st]; __syncthreads(); }
    const float rms = rsqrtf(red[0] * (1.0f / D_MODEL) + 1e-6f);
    __syncthreads();
    float sc = 0.f;
#pragma unroll
    for (int j = 0; j < 4; ++j) {
        int d = tid + j * 256;
        float h = xv[j] * rms * norm_w[d];
        h_bf[(size_t)row * D_MODEL + d] = f2bf(h);
        sc += h * scorer_w[d];
    }
    red[tid] = sc; __syncthreads();
    for (int st = 128; st > 0; st >>= 1) { if (tid < st) red[tid] += red[tid + st]; __syncthreads(); }
    if (tid == 0) scores[row] = red[0] + scorer_b[0];
}

// ---------------- Kernel 2: top-256 selection ----------------
__global__ __launch_bounds__(1024)
void topk_kernel(const float* __restrict__ scores, int* __restrict__ idx_out) {
    __shared__ float sc[S_TOK];
    __shared__ float lmax[1024];
    __shared__ int   lidx[1024];
    const int tid = threadIdx.x;
    for (int i = tid; i < S_TOK; i += 1024) sc[i] = scores[i];
    __syncthreads();
    for (int it = 0; it < G_TOK; ++it) {
        float bm = NEGINF; int bi = 0x7fffffff;
#pragma unroll
        for (int j = 0; j < 8; ++j) {
            int i = tid + j * 1024;
            float v = sc[i];
            if (v > bm || (v == bm && i < bi)) { bm = v; bi = i; }
        }
        lmax[tid] = bm; lidx[tid] = bi; __syncthreads();
        for (int st = 512; st > 0; st >>= 1) {
            if (tid < st) {
                float vo = lmax[tid + st]; int io = lidx[tid + st];
                if (vo > lmax[tid] || (vo == lmax[tid] && io < lidx[tid])) { lmax[tid] = vo; lidx[tid] = io; }
            }
            __syncthreads();
        }
        if (tid == 0) { idx_out[it] = lidx[0]; sc[lidx[0]] = NEGINF; }
        __syncthreads();
    }
}

// ---------------- Kernel 3: transpose + fp32->bf16 weight convert ----------------
__global__ __launch_bounds__(256)
void convw_kernel(const float* __restrict__ src, unsigned short* __restrict__ dst, int R, int C) {
    int i = blockIdx.x * 256 + threadIdx.x;   // exact grid
    int r = i / C, c = i % C;
    dst[(size_t)c * R + r] = f2bf(src[i]);
}

// ---------------- Kernel 4/7: bf16 WMMA GEMM (A row-major, B supplied transposed) ----------------
// block: 128 threads (4 waves); block tile 128(M) x 64(N); K in steps of 32.
// mode 0: bf16 out as [H][S][64];  mode 1: bf16 out as [H][64][S];  mode 2: f32 out = resid + acc.
__global__ __launch_bounds__(128)
void gemm_bf16_kernel(const unsigned short* __restrict__ A, const unsigned short* __restrict__ Bt,
                      int K, int N, unsigned short* __restrict__ dst_bf,
                      float* __restrict__ dst_f, const float* __restrict__ resid,
                      int mode, int Stot) {
    const int wave = threadIdx.x >> 5, lane = threadIdx.x & 31;
    const int half = lane >> 4, ln = lane & 15;
    const int tile_m = blockIdx.x * 128, tile_n = blockIdx.y * 64;
    const int rm0 = tile_m + wave * 32;
    v8f zero = {};
    v8f c[2][4];
#pragma unroll
    for (int i = 0; i < 2; ++i)
#pragma unroll
        for (int t = 0; t < 4; ++t) c[i][t] = zero;

    for (int k = 0; k < K; k += 32) {
        v16bf a0 = load_frag_g(A, K, rm0, k);
        v16bf a1 = load_frag_g(A, K, rm0 + 16, k);
#pragma unroll
        for (int t = 0; t < 4; ++t) {
            v16bf b = load_frag_g(Bt, K, tile_n + t * 16, k);
            c[0][t] = __builtin_amdgcn_wmma_f32_16x16x32_bf16(false, a0, false, b, (short)0, c[0][t], false, false);
            c[1][t] = __builtin_amdgcn_wmma_f32_16x16x32_bf16(false, a1, false, b, (short)0, c[1][t], false, false);
        }
    }
#pragma unroll
    for (int i = 0; i < 2; ++i)
#pragma unroll
        for (int t = 0; t < 4; ++t)
#pragma unroll
            for (int r = 0; r < 8; ++r) {
                int row = rm0 + i * 16 + r + half * 8;
                int col = tile_n + t * 16 + ln;
                float v = c[i][t][r];
                if (mode == 2) {
                    size_t o = (size_t)row * N + col;
                    dst_f[o] = resid[o] + v;
                } else if (mode == 1) {
                    dst_bf[((size_t)(col >> 6) * 64 + (col & 63)) * Stot + row] = f2bf(v);
                } else {
                    dst_bf[((size_t)(col >> 6) * Stot + row) * 64 + (col & 63)] = f2bf(v);
                }
            }
}

// ---------------- Kernel 5: gather global K/V ----------------
__global__ __launch_bounds__(64)
void gather_kernel(const unsigned short* __restrict__ k_bf, const unsigned short* __restrict__ vt,
                   const int* __restrict__ idx, unsigned short* __restrict__ kg,
                   unsigned short* __restrict__ vgt) {
    const int g = blockIdx.x, head = blockIdx.y, hd = threadIdx.x;
    const int s = idx[g];
    kg[((size_t)head * G_TOK + g) * HD + hd] = k_bf[((size_t)head * S_TOK + s) * HD + hd];
    vgt[((size_t)head * HD + hd) * G_TOK + g] = vt[((size_t)head * HD + hd) * S_TOK + s];
}

// ---------------- Kernel 6: windowed-causal + global flash attention ----------------
// grid (S/64, H); 4 waves, each owns 16 query rows. Joint online softmax over
// local window keys (causal) and 256 gathered global keys (idx[g] <= qpos).
__global__ __launch_bounds__(128)
void attn_kernel(const unsigned short* __restrict__ q_bf, const unsigned short* __restrict__ k_bf,
                 const unsigned short* __restrict__ vt, const unsigned short* __restrict__ kg,
                 const unsigned short* __restrict__ vgt, const int* __restrict__ idx,
                 unsigned short* __restrict__ o_bf) {
    __shared__ unsigned short p_sh[4][16][32];
    const int wave = threadIdx.x >> 5, lane = threadIdx.x & 31;
    const int half = lane >> 4, ln = lane & 15;
    const int head = blockIdx.y;
    const int tile_m = blockIdx.x * 64;
    const int qr0 = tile_m + wave * 16;
    const float scale = 0.125f;   // 1/sqrt(64)

    const unsigned short* Qh = q_bf + (size_t)head * S_TOK * HD;
    const unsigned short* Kh = k_bf + (size_t)head * S_TOK * HD;
    const unsigned short* Vh = vt + (size_t)head * HD * S_TOK;
    const unsigned short* Kg = kg + (size_t)head * G_TOK * HD;
    const unsigned short* Vg = vgt + (size_t)head * HD * G_TOK;

    const v16bf qa0 = load_frag_g(Qh, HD, qr0, 0);
    const v16bf qa1 = load_frag_g(Qh, HD, qr0, 32);

    v8f zero = {};
    v8f o0 = zero, o1 = zero, o2 = zero, o3 = zero;
    float m_s[8], l_s[8];
#pragma unroll
    for (int r = 0; r < 8; ++r) { m_s[r] = NEGINF; l_s[r] = 0.f; }

    const int win0 = (tile_m / WIN) * WIN;
    const int n_local = (qr0 + 16 - win0 + 31) / 32;
    const int n_total = n_local + G_TOK / 32;

    for (int ch = 0; ch < n_total; ++ch) {
        const bool is_local = ch < n_local;
        const int kc = is_local ? (win0 + ch * 32) : ((ch - n_local) * 32);

        v8f s0 = zero, s1 = zero;
        if (is_local) {
            v16bf b;
            b = load_frag_g(Kh, HD, kc, 0);
            s0 = __builtin_amdgcn_wmma_f32_16x16x32_bf16(false, qa0, false, b, (short)0, s0, false, false);
            b = load_frag_g(Kh, HD, kc, 32);
            s0 = __builtin_amdgcn_wmma_f32_16x16x32_bf16(false, qa1, false, b, (short)0, s0, false, false);
            b = load_frag_g(Kh, HD, kc + 16, 0);
            s1 = __builtin_amdgcn_wmma_f32_16x16x32_bf16(false, qa0, false, b, (short)0, s1, false, false);
            b = load_frag_g(Kh, HD, kc + 16, 32);
            s1 = __builtin_amdgcn_wmma_f32_16x16x32_bf16(false, qa1, false, b, (short)0, s1, false, false);
        } else {
            v16bf b;
            b = load_frag_g(Kg, HD, kc, 0);
            s0 = __builtin_amdgcn_wmma_f32_16x16x32_bf16(false, qa0, false, b, (short)0, s0, false, false);
            b = load_frag_g(Kg, HD, kc, 32);
            s0 = __builtin_amdgcn_wmma_f32_16x16x32_bf16(false, qa1, false, b, (short)0, s0, false, false);
            b = load_frag_g(Kg, HD, kc + 16, 0);
            s1 = __builtin_amdgcn_wmma_f32_16x16x32_bf16(false, qa0, false, b, (short)0, s1, false, false);
            b = load_frag_g(Kg, HD, kc + 16, 32);
            s1 = __builtin_amdgcn_wmma_f32_16x16x32_bf16(false, qa1, false, b, (short)0, s1, false, false);
        }

        // key position for this lane's column (same column for both halves)
        int kpos0, kpos1;
        if (is_local) { kpos0 = kc + ln; kpos1 = kc + 16 + ln; }
        else          { kpos0 = idx[kc + ln]; kpos1 = idx[kc + 16 + ln]; }

        float alpha[8];
#pragma unroll
        for (int r = 0; r < 8; ++r) {
            const int qpos = qr0 + r + half * 8;
            float a = (kpos0 <= qpos) ? s0[r] * scale : NEGINF;
            float bm = (kpos1 <= qpos) ? s1[r] * scale : NEGINF;
            // row max across the 16 lanes of this half-wave
            float mx = fmaxf(a, bm);
#pragma unroll
            for (int off = 8; off >= 1; off >>= 1) mx = fmaxf(mx, __shfl_xor(mx, off, 16));
            float mnew = fmaxf(m_s[r], mx);
            alpha[r] = __expf(m_s[r] - mnew);
            m_s[r] = mnew;
            float p0 = __expf(a - mnew), p1 = __expf(bm - mnew);
            float rs = p0 + p1;
#pragma unroll
            for (int off = 8; off >= 1; off >>= 1) rs += __shfl_xor(rs, off, 16);
            l_s[r] = l_s[r] * alpha[r] + rs;
            s0[r] = p0; s1[r] = p1;
        }
#pragma unroll
        for (int r = 0; r < 8; ++r) { o0[r] *= alpha[r]; o1[r] *= alpha[r]; o2[r] *= alpha[r]; o3[r] *= alpha[r]; }

        // C-layout -> A-layout reshape of P through per-wave LDS tile (bf16)
#pragma unroll
        for (int r = 0; r < 8; ++r) {
            p_sh[wave][r + half * 8][ln] = f2bf(s0[r]);
            p_sh[wave][r + half * 8][16 + ln] = f2bf(s1[r]);
        }
        // per-wave LDS is in-order: reads below observe the writes above
        union { uint4 q[2]; v16bf v; } pu;
        const unsigned short* rowp = &p_sh[wave][ln][0];
        const int koff = half * 8;
        pu.q[0] = *(const uint4*)(rowp + koff);
        pu.q[1] = *(const uint4*)(rowp + koff + 16);
        const v16bf pa = pu.v;

        if (is_local) {
            v16bf b;
            b = load_frag_g(Vh, S_TOK, 0, kc);
            o0 = __builtin_amdgcn_wmma_f32_16x16x32_bf16(false, pa, false, b, (short)0, o0, false, false);
            b = load_frag_g(Vh, S_TOK, 16, kc);
            o1 = __builtin_amdgcn_wmma_f32_16x16x32_bf16(false, pa, false, b, (short)0, o1, false, false);
            b = load_frag_g(Vh, S_TOK, 32, kc);
            o2 = __builtin_amdgcn_wmma_f32_16x16x32_bf16(false, pa, false, b, (short)0, o2, false, false);
            b = load_frag_g(Vh, S_TOK, 48, kc);
            o3 = __builtin_amdgcn_wmma_f32_16x16x32_bf16(false, pa, false, b, (short)0, o3, false, false);
        } else {
            v16bf b;
            b = load_frag_g(Vg, G_TOK, 0, kc);
            o0 = __builtin_amdgcn_wmma_f32_16x16x32_bf16(false, pa, false, b, (short)0, o0, false, false);
            b = load_frag_g(Vg, G_TOK, 16, kc);
            o1 = __builtin_amdgcn_wmma_f32_16x16x32_bf16(false, pa, false, b, (short)0, o1, false, false);
            b = load_frag_g(Vg, G_TOK, 32, kc);
            o2 = __builtin_amdgcn_wmma_f32_16x16x32_bf16(false, pa, false, b, (short)0, o2, false, false);
            b = load_frag_g(Vg, G_TOK, 48, kc);
            o3 = __builtin_amdgcn_wmma_f32_16x16x32_bf16(false, pa, false, b, (short)0, o3, false, false);
        }
    }

    // normalize and store o as [S][H*HD] bf16
#pragma unroll
    for (int r = 0; r < 8; ++r) {
        const float inv = 1.0f / l_s[r];
        const int row = qr0 + r + half * 8;
        const size_t base = (size_t)row * D_MODEL + head * HD;
        o_bf[base + 0 * 16 + ln] = f2bf(o0[r] * inv);
        o_bf[base + 1 * 16 + ln] = f2bf(o1[r] * inv);
        o_bf[base + 2 * 16 + ln] = f2bf(o2[r] * inv);
        o_bf[base + 3 * 16 + ln] = f2bf(o3[r] * inv);
    }
}

// ---------------- launch ----------------
extern "C" void kernel_launch(void* const* d_in, const int* in_sizes, int n_in,
                              void* d_out, int out_size, void* d_ws, size_t ws_size,
                              hipStream_t stream) {
    const float* x        = (const float*)d_in[0];
    const float* norm_w   = (const float*)d_in[1];
    const float* w_q      = (const float*)d_in[2];
    const float* w_k      = (const float*)d_in[3];
    const float* w_v      = (const float*)d_in[4];
    const float* w_out    = (const float*)d_in[5];
    const float* scorer_w = (const float*)d_in[6];
    const float* scorer_b = (const float*)d_in[7];
    float* out = (float*)d_out;

    char* ws = (char*)d_ws;
    size_t off = 0;
    auto alloc = [&](size_t bytes) { size_t o = off; off += (bytes + 255) & ~(size_t)255; return o; };

    unsigned short* h_bf  = (unsigned short*)(ws + alloc((size_t)S_TOK * D_MODEL * 2));
    float*          score = (float*)         (ws + alloc((size_t)S_TOK * 4));
    int*            idx   = (int*)           (ws + alloc((size_t)G_TOK * 4));
    unsigned short* wq_t  = (unsigned short*)(ws + alloc((size_t)D_MODEL * D_MODEL * 2));
    unsigned short* wk_t  = (unsigned short*)(ws + alloc((size_t)D_MODEL * D_MODEL * 2));
    unsigned short* wv_t  = (unsigned short*)(ws + alloc((size_t)D_MODEL * D_MODEL * 2));
    unsigned short* wo_t  = (unsigned short*)(ws + alloc((size_t)D_MODEL * D_MODEL * 2));
    unsigned short* q_bf  = (unsigned short*)(ws + alloc((size_t)NHEAD * S_TOK * HD * 2));
    unsigned short* k_bf  = (unsigned short*)(ws + alloc((size_t)NHEAD * S_TOK * HD * 2));
    unsigned short* v_t   = (unsigned short*)(ws + alloc((size_t)NHEAD * HD * S_TOK * 2));
    unsigned short* kg    = (unsigned short*)(ws + alloc((size_t)NHEAD * G_TOK * HD * 2));
    unsigned short* vgt   = (unsigned short*)(ws + alloc((size_t)NHEAD * HD * G_TOK * 2));
    unsigned short* o_bf  = (unsigned short*)(ws + alloc((size_t)S_TOK * D_MODEL * 2));
    (void)ws_size; (void)in_sizes; (void)n_in; (void)out_size;

    rmsnorm_score_kernel<<<S_TOK, 256, 0, stream>>>(x, norm_w, scorer_w, scorer_b, h_bf, score);
    topk_kernel<<<1, 1024, 0, stream>>>(score, idx);

    const int wblk = D_MODEL * D_MODEL / 256;
    convw_kernel<<<wblk, 256, 0, stream>>>(w_q, wq_t, D_MODEL, D_MODEL);
    convw_kernel<<<wblk, 256, 0, stream>>>(w_k, wk_t, D_MODEL, D_MODEL);
    convw_kernel<<<wblk, 256, 0, stream>>>(w_v, wv_t, D_MODEL, D_MODEL);
    convw_kernel<<<wblk, 256, 0, stream>>>(w_out, wo_t, D_MODEL, D_MODEL);

    dim3 ggrid(S_TOK / 128, D_MODEL / 64);
    gemm_bf16_kernel<<<ggrid, 128, 0, stream>>>(h_bf, wq_t, D_MODEL, D_MODEL, q_bf, nullptr, nullptr, 0, S_TOK);
    gemm_bf16_kernel<<<ggrid, 128, 0, stream>>>(h_bf, wk_t, D_MODEL, D_MODEL, k_bf, nullptr, nullptr, 0, S_TOK);
    gemm_bf16_kernel<<<ggrid, 128, 0, stream>>>(h_bf, wv_t, D_MODEL, D_MODEL, v_t, nullptr, nullptr, 1, S_TOK);

    gather_kernel<<<dim3(G_TOK, NHEAD), 64, 0, stream>>>(k_bf, v_t, idx, kg, vgt);

    attn_kernel<<<dim3(S_TOK / 64, NHEAD), 128, 0, stream>>>(q_bf, k_bf, v_t, kg, vgt, idx, o_bf);

    gemm_bf16_kernel<<<ggrid, 128, 0, stream>>>(o_bf, wo_t, D_MODEL, D_MODEL, nullptr, out, x, 2, S_TOK);
}
